// GCNNet_29033978921579
// MI455X (gfx1250) — matrix operand order repaired and checked by reading, not compiled, not run on hardware
//
#include <hip/hip_runtime.h>
#include <hip/hip_bf16.h>

// ---------------------------------------------------------------------------
// Types for CDNA5 WMMA (wave32): v_wmma_f32_16x16x32_bf16
// ---------------------------------------------------------------------------
typedef __attribute__((ext_vector_type(16))) __bf16 v16bf;
typedef __attribute__((ext_vector_type(8)))  __bf16 v8bf;
typedef __attribute__((ext_vector_type(8)))  float  v8f;

#define B_GRAPHS 8192
#define NPG 50
#define EPG 100
#define F_XD 78
#define OUT_DIM 128

// Fused drug-branch kernel LDS layout (dynamic LDS; 186 KB < 320 KB/WGP)
#define KPMAX 160          // max padded K over the 3 conv layers (156 -> 160)
#define NPMAX 320          // max padded N over the 3 conv layers (312 -> 320)
#define SM_XB_BYTES (64 * KPMAX * 2)                 // bf16 activations, 20480
#define SM_HB_OFF   (SM_XB_BYTES)                    // f32 h = x@W, 81920
#define SM_AB_OFF   (SM_HB_OFF + 64 * NPMAX * 4)     // f32 aggregation, 81920
#define SM_DINV_OFF (SM_AB_OFF + 64 * NPMAX * 4)     // f32[64]
#define SM_NORM_OFF (SM_DINV_OFF + 64 * 4)           // f32[100]
#define SM_SRC_OFF  (SM_NORM_OFF + EPG * 4)          // i32[100]
#define SM_DST_OFF  (SM_SRC_OFF + EPG * 4)           // i32[100]
#define SMEM_TOTAL  (SM_DST_OFF + EPG * 4)           // 185776 bytes

// ---------------------------------------------------------------------------
// A-fragment (16-bit A, 16x32): lane holds row (lane%16); element j maps to
//   K = half*16 + j           half = lane/16
// -> one contiguous 16-element (32-byte) run.
// ---------------------------------------------------------------------------
__device__ __forceinline__ v16bf a_frag(const __bf16* p, int half) {
  return *(const v16bf*)(p + half * 16);
}

// ---------------------------------------------------------------------------
// One GCNConv layer entirely in LDS for a single 50-node graph.
//   xb : bf16 [64 x Kp] input activations (rows 50..63 zero)
//   hb : f32  [64 x Np] h = x @ W           (WMMA output)
//   ab : f32  [64 x Np] deg-normalized aggregation + bias + relu
// W is padded column-major bf16 [Np x Kp] so a B-fragment is one 32B read.
// ---------------------------------------------------------------------------
__device__ __forceinline__ void gcn_layer(
    __bf16* xb, float* hb, float* ab,
    const float* dinv, const float* norm, const int* srcl, const int* dstl,
    const __bf16* __restrict__ Wcm, const float* __restrict__ bias,
    int Kp, int N, int Np, int Kpn, int last, int tid)
{
  const int wave = tid >> 5;
  const int lane = tid & 31;
  const int half = lane >> 4;
  const int lr   = lane & 15;

  // --- GEMM: 4 M-tiles x (Np/16) N-tiles, distributed over 8 waves ---------
  const int tiles = 4 * (Np >> 4);
  for (int t = wave; t < tiles; t += 8) {
    const int mt = t & 3;
    const int nt = t >> 2;
    v8f acc;
#pragma unroll
    for (int i = 0; i < 8; ++i) acc[i] = 0.0f;
    const __bf16* arow = xb + (mt * 16 + lr) * Kp;
    const __bf16* bcol = Wcm + (nt * 16 + lr) * Kp + half * 16;
    for (int k0 = 0; k0 < Kp; k0 += 32) {
      v16bf a = a_frag(arow + k0, half);
      v16bf b = *(const v16bf*)(bcol + k0);   // 16 consecutive K, 32B aligned
      acc = __builtin_amdgcn_wmma_f32_16x16x32_bf16(
          false, a, false, b, (short)0, acc, false, false);
    }
    const int ncol = nt * 16 + lr;
#pragma unroll
    for (int i = 0; i < 8; ++i)
      hb[(mt * 16 + half * 8 + i) * Np + ncol] = acc[i];   // C/D layout
  }
  __syncthreads();

  // --- neighborhood aggregation with LDS float atomics ---------------------
  for (int i = tid; i < 64 * Np; i += 256) ab[i] = 0.0f;
  __syncthreads();
  for (int i = tid; i < EPG * N; i += 256) {
    int e = i / N, f = i - e * N;
    atomicAdd(&ab[dstl[e] * Np + f], hb[srcl[e] * Np + f] * norm[e]);
  }
  __syncthreads();

  // --- self-loop term + bias + relu ----------------------------------------
  for (int i = tid; i < NPG * N; i += 256) {
    int r = i / N, f = i - r * N;
    float v = ab[r * Np + f] + hb[r * Np + f] * dinv[r] * dinv[r] + bias[f];
    ab[r * Np + f] = fmaxf(v, 0.0f);
  }
  __syncthreads();

  // --- repack as bf16 input for the next layer -----------------------------
  if (!last) {
    for (int i = tid; i < 64 * Kpn; i += 256) {
      int r = i / Kpn, k = i - r * Kpn;
      float v = (r < NPG && k < N) ? ab[r * Np + k] : 0.0f;
      xb[i] = (__bf16)v;
    }
    __syncthreads();
  }
}

// ---------------------------------------------------------------------------
// Fused drug branch: degrees + 3 GCNConv layers + max-pool, one block/graph.
// ---------------------------------------------------------------------------
__global__ __launch_bounds__(256, 1) void gcn_drug_kernel(
    const float* __restrict__ x, const long long* __restrict__ ei,
    const __bf16* __restrict__ W1, const float* __restrict__ b1,
    const __bf16* __restrict__ W2, const float* __restrict__ b2,
    const __bf16* __restrict__ W3, const float* __restrict__ b3,
    float* __restrict__ gout)
{
  extern __shared__ char smem[];
  __bf16* xb  = (__bf16*)smem;
  float* hb   = (float*)(smem + SM_HB_OFF);
  float* ab   = (float*)(smem + SM_AB_OFF);
  float* dinv = (float*)(smem + SM_DINV_OFF);
  float* norm = (float*)(smem + SM_NORM_OFF);
  int* srcl   = (int*)(smem + SM_SRC_OFF);
  int* dstl   = (int*)(smem + SM_DST_OFF);

  const int g   = blockIdx.x;
  const int tid = threadIdx.x;
  const long long nbase = (long long)g * NPG;
  const long long ebase = (long long)g * EPG;
  const long long Etot  = (long long)B_GRAPHS * EPG;

  // degrees (edges of a graph are contiguous and graph-local)
  if (tid < 64) dinv[tid] = 1.0f;     // +1 self loop
  __syncthreads();
  if (tid < EPG) {
    int s = (int)(ei[ebase + tid] - nbase);          // edge_index row 0: src
    int d = (int)(ei[Etot + ebase + tid] - nbase);   // edge_index row 1: dst
    srcl[tid] = s;
    dstl[tid] = d;
    atomicAdd(&dinv[d], 1.0f);
  }
  __syncthreads();
  if (tid < 64) dinv[tid] = rsqrtf(dinv[tid]);
  __syncthreads();
  if (tid < EPG) norm[tid] = dinv[srcl[tid]] * dinv[dstl[tid]];

  // load node features (f32 -> bf16, padded to 64 x 96)
  for (int i = tid; i < 64 * 96; i += 256) {
    int r = i / 96, k = i - r * 96;
    float v = (r < NPG && k < F_XD) ? x[(nbase + r) * F_XD + k] : 0.0f;
    xb[i] = (__bf16)v;
  }
  __syncthreads();

  gcn_layer(xb, hb, ab, dinv, norm, srcl, dstl, W1, b1,  96,  78,  80,  96, 0, tid);
  gcn_layer(xb, hb, ab, dinv, norm, srcl, dstl, W2, b2,  96, 156, 160, 160, 0, tid);
  gcn_layer(xb, hb, ab, dinv, norm, srcl, dstl, W3, b3, 160, 312, 320,   0, 1, tid);

  // global max pool over the 50 nodes
  for (int f = tid; f < 312; f += 256) {
    float m = -3.402823466e38f;
    for (int r = 0; r < NPG; ++r) m = fmaxf(m, ab[r * 320 + f]);
    gout[(long long)g * 312 + f] = m;
  }
}

// ---------------------------------------------------------------------------
// Conv-weight prep: f32 row-major [K,N] -> bf16 padded column-major [Np,Kp]
// ---------------------------------------------------------------------------
__global__ void prep_w_kernel(const float* __restrict__ W, __bf16* __restrict__ out,
                              int K, int N, int Kp, int Np)
{
  int i = blockIdx.x * 256 + threadIdx.x;
  if (i >= Kp * Np) return;
  int n = i / Kp, k = i - n * Kp;
  out[i] = (__bf16)((n < N && k < K) ? W[k * N + n] : 0.0f);
}

// ---------------------------------------------------------------------------
// Generic dense GEMM: C[m, n] = act(A[M,K] @ W[K,N] + bias), WMMA bf16.
// 4 waves/block, 64x64 block tile; A/W staged to LDS with f32->bf16 convert.
// ---------------------------------------------------------------------------
__global__ __launch_bounds__(128) void gemm_kernel(
    const float* __restrict__ A, int lda,
    const float* __restrict__ W, const float* __restrict__ bias,
    float* __restrict__ C, int ldc,
    int M, int K, int N, int relu)
{
  __shared__ __bf16 At[64 * 32];   // [m][k], k contiguous
  __shared__ __bf16 Bt[64 * 32];   // [n][k], k contiguous (column-major tile)

  const int tid  = threadIdx.x;
  const int wave = tid >> 5;
  const int lane = tid & 31;
  const int half = lane >> 4;
  const int lr   = lane & 15;
  const int m0 = blockIdx.y * 64;
  const int n0 = blockIdx.x * 64;

  v8f acc[4];
#pragma unroll
  for (int mt = 0; mt < 4; ++mt)
#pragma unroll
    for (int i = 0; i < 8; ++i) acc[mt][i] = 0.0f;

  for (int k0 = 0; k0 < K; k0 += 32) {
    __syncthreads();
    for (int i = tid; i < 64 * 32; i += 128) {
      int r = i >> 5, k = i & 31;
      int gk = k0 + k;
      int gm = m0 + r;
      At[i] = (__bf16)((gm < M && gk < K) ? A[(long long)gm * lda + gk] : 0.0f);
      int gn = n0 + r;
      Bt[i] = (__bf16)((gn < N && gk < K) ? W[(long long)gk * N + gn] : 0.0f);
    }
    __syncthreads();
    const __bf16* bptr = Bt + (wave * 16 + lr) * 32 + half * 16;
    v16bf b = *(const v16bf*)bptr;
#pragma unroll
    for (int mt = 0; mt < 4; ++mt) {
      const __bf16* arow = At + (mt * 16 + lr) * 32;
      v16bf a = a_frag(arow, half);
      acc[mt] = __builtin_amdgcn_wmma_f32_16x16x32_bf16(
          false, a, false, b, (short)0, acc[mt], false, false);
    }
  }

#pragma unroll
  for (int mt = 0; mt < 4; ++mt) {
#pragma unroll
    for (int i = 0; i < 8; ++i) {
      int m = m0 + mt * 16 + half * 8 + i;
      int n = n0 + wave * 16 + lr;
      if (m < M && n < N) {
        float v = acc[mt][i] + bias[n];
        if (relu) v = fmaxf(v, 0.0f);
        C[(long long)m * ldc + n] = v;
      }
    }
  }
}

// ---------------------------------------------------------------------------
// Row-wise L2 normalize for the cell-line features [8192, 954]
// ---------------------------------------------------------------------------
__global__ __launch_bounds__(256) void normalize_kernel(
    const float* __restrict__ cell, float* __restrict__ cn)
{
  __shared__ float red[256];
  const int r = blockIdx.x, tid = threadIdx.x;
  const float* row = cell + (long long)r * 954;
  float s = 0.0f;
  for (int k = tid; k < 954; k += 256) { float v = row[k]; s += v * v; }
  red[tid] = s;
  __syncthreads();
  for (int off = 128; off > 0; off >>= 1) {
    if (tid < off) red[tid] += red[tid + off];
    __syncthreads();
  }
  float scale = 1.0f / fmaxf(sqrtf(red[0]), 1e-12f);
  for (int k = tid; k < 954; k += 256)
    cn[(long long)r * 954 + k] = row[k] * scale;
}

// ---------------------------------------------------------------------------
// Host orchestration
// ---------------------------------------------------------------------------
static inline void* ws_alloc(char* ws, size_t& off, size_t bytes) {
  off = (off + 255) & ~size_t(255);
  void* p = ws + off;
  off += bytes;
  return p;
}

extern "C" void kernel_launch(void* const* d_in, const int* in_sizes, int n_in,
                              void* d_out, int out_size, void* d_ws, size_t ws_size,
                              hipStream_t stream)
{
  (void)in_sizes; (void)n_in; (void)out_size; (void)ws_size;

  const float*     x1   = (const float*)d_in[0];
  const long long* ei1  = (const long long*)d_in[1];
  const float*     x2   = (const float*)d_in[3];
  const long long* ei2  = (const long long*)d_in[4];
  const float*     cell = (const float*)d_in[6];
  const float* Wc1 = (const float*)d_in[7];  const float* bc1 = (const float*)d_in[8];
  const float* Wc2 = (const float*)d_in[9];  const float* bc2 = (const float*)d_in[10];
  const float* Wc3 = (const float*)d_in[11]; const float* bc3 = (const float*)d_in[12];
  const float* Wg1 = (const float*)d_in[13]; const float* bg1 = (const float*)d_in[14];
  const float* Wg2 = (const float*)d_in[15]; const float* bg2 = (const float*)d_in[16];
  const float* Wr1 = (const float*)d_in[17]; const float* br1 = (const float*)d_in[18];
  const float* Wr2 = (const float*)d_in[19]; const float* br2 = (const float*)d_in[20];
  const float* Wr3 = (const float*)d_in[21]; const float* br3 = (const float*)d_in[22];
  const float* Wf1 = (const float*)d_in[23]; const float* bf1 = (const float*)d_in[24];
  const float* Wf2 = (const float*)d_in[25]; const float* bf2 = (const float*)d_in[26];
  const float* Wo  = (const float*)d_in[27]; const float* bo  = (const float*)d_in[28];
  float* out = (float*)d_out;

  char* ws = (char*)d_ws;
  size_t off = 0;
  __bf16* W1cm = (__bf16*)ws_alloc(ws, off, (size_t)80  * 96  * 2);
  __bf16* W2cm = (__bf16*)ws_alloc(ws, off, (size_t)160 * 96  * 2);
  __bf16* W3cm = (__bf16*)ws_alloc(ws, off, (size_t)320 * 160 * 2);
  float* g1 = (float*)ws_alloc(ws, off, (size_t)B_GRAPHS * 312 * 4);
  float* g2 = (float*)ws_alloc(ws, off, (size_t)B_GRAPHS * 312 * 4);
  float* cn = (float*)ws_alloc(ws, off, (size_t)B_GRAPHS * 954 * 4);
  float* t1 = (float*)ws_alloc(ws, off, (size_t)B_GRAPHS * 512 * 4);
  float* t2 = (float*)ws_alloc(ws, off, (size_t)B_GRAPHS * 512 * 4);
  float* xc = (float*)ws_alloc(ws, off, (size_t)B_GRAPHS * 384 * 4);

  // --- conv-weight prep (bf16, padded column-major) -------------------------
  prep_w_kernel<<<(80 * 96 + 255) / 256, 256, 0, stream>>>(Wc1, W1cm, 78, 78, 96, 80);
  prep_w_kernel<<<(160 * 96 + 255) / 256, 256, 0, stream>>>(Wc2, W2cm, 78, 156, 96, 160);
  prep_w_kernel<<<(320 * 160 + 255) / 256, 256, 0, stream>>>(Wc3, W3cm, 156, 312, 160, 320);

  // --- fused drug branches (needs >64KB dynamic LDS: 186 KB / 320 KB WGP) ---
  hipFuncSetAttribute((const void*)gcn_drug_kernel,
                      hipFuncAttributeMaxDynamicSharedMemorySize, SMEM_TOTAL);
  gcn_drug_kernel<<<B_GRAPHS, 256, SMEM_TOTAL, stream>>>(
      x1, ei1, W1cm, bc1, W2cm, bc2, W3cm, bc3, g1);
  gcn_drug_kernel<<<B_GRAPHS, 256, SMEM_TOTAL, stream>>>(
      x2, ei2, W1cm, bc1, W2cm, bc2, W3cm, bc3, g2);

  const int M = B_GRAPHS;
  const dim3 blk(128);
  auto grid = [](int n, int m) { return dim3((unsigned)((n + 63) / 64),
                                             (unsigned)((m + 63) / 64)); };

  // --- drug heads: g -> relu(312x156) -> 156x128, written into concat buf ---
  gemm_kernel<<<grid(156, M), blk, 0, stream>>>(g1, 312, Wg1, bg1, t1, 156, M, 312, 156, 1);
  gemm_kernel<<<grid(128, M), blk, 0, stream>>>(t1, 156, Wg2, bg2, xc + 0, 384, M, 156, 128, 0);
  gemm_kernel<<<grid(156, M), blk, 0, stream>>>(g2, 312, Wg1, bg1, t1, 156, M, 312, 156, 1);
  gemm_kernel<<<grid(128, M), blk, 0, stream>>>(t1, 156, Wg2, bg2, xc + 128, 384, M, 156, 128, 0);

  // --- cell branch ----------------------------------------------------------
  normalize_kernel<<<B_GRAPHS, 256, 0, stream>>>(cell, cn);
  gemm_kernel<<<grid(512, M), blk, 0, stream>>>(cn, 954, Wr1, br1, t1, 512, M, 954, 512, 1);
  gemm_kernel<<<grid(256, M), blk, 0, stream>>>(t1, 512, Wr2, br2, t2, 256, M, 512, 256, 1);
  gemm_kernel<<<grid(128, M), blk, 0, stream>>>(t2, 256, Wr3, br3, xc + 256, 384, M, 256, 128, 0);

  // --- fusion head ----------------------------------------------------------
  gemm_kernel<<<grid(256, M), blk, 0, stream>>>(xc, 384, Wf1, bf1, t1, 256, M, 384, 256, 1);
  gemm_kernel<<<grid(128, M), blk, 0, stream>>>(t1, 256, Wf2, bf2, t2, 128, M, 256, 128, 1);
  gemm_kernel<<<grid(2, M),   blk, 0, stream>>>(t2, 128, Wo, bo, out, 2, M, 128, 2, 0);
}
